// Kmeans_20675972563185
// MI455X (gfx1250) — matrix operand order repaired and checked by reading, not compiled, not run on hardware
//
#include <hip/hip_runtime.h>
#include <hip/hip_bf16.h>

// Sizes fixed by the reference problem.
#define NPTS 131072
#define DDIM 768
#define KCEN 2048
#define KCHUNKS 24      // 768 / 32
#define COLTILES 128    // 2048 / 16
#define WAVES_PER_BLOCK 4

typedef __attribute__((ext_vector_type(16))) __bf16 v16bf;
typedef __attribute__((ext_vector_type(8)))  float  v8f;

// ---------------------------------------------------------------------------
// Pass 1: centroids [D][K] f32  ->  transposed bf16 [K][D] in workspace.
// Each WMMA B-fragment (16 consecutive K-dim values for one centroid column)
// becomes a contiguous 32-byte load.
// ---------------------------------------------------------------------------
__global__ void kmeans_convertT_kernel(const float* __restrict__ c,
                                       __bf16* __restrict__ ct) {
    int idx = blockIdx.x * 256 + threadIdx.x;     // index into ct[k][d]
    if (idx >= KCEN * DDIM) return;
    int k = idx / DDIM;
    int d = idx - k * DDIM;
    ct[idx] = (__bf16)c[(size_t)d * KCEN + k];
}

// ---------------------------------------------------------------------------
// Pass 2: fused GEMM + argmin. One wave handles a 16-row tile of x.
// The 16x768 A tile is converted to bf16 once into a VGPR-resident fragment
// array (24 x v16bf = 192 VGPRs), then we sweep all 128 centroid column
// tiles with v_wmma_f32_16x16x32_bf16 using two independent accumulator
// chains, fusing the distance epilogue + running argmin so the [N,K]
// distance matrix is never materialized.
// ---------------------------------------------------------------------------
__global__ __launch_bounds__(WAVES_PER_BLOCK * 32)
void kmeans_argmin_kernel(const float* __restrict__ x,
                          const __bf16* __restrict__ ct,     // [K][D] bf16
                          const float* __restrict__ cnorm,   // [K]
                          float* __restrict__ out) {
    const int wave = threadIdx.x >> 5;
    const int lane = threadIdx.x & 31;
    const int r     = lane & 15;   // row within tile (A) / column within tile (B,D)
    const int khalf = lane >> 4;   // which K half of the fragment this lane holds

    const int rowTile = blockIdx.x * WAVES_PER_BLOCK + wave;
    const int row0 = rowTile * 16;

    // -------- Load + convert A tile straight into register fragments --------
    // Fragment element i (16-bit A 16x32 layout, contiguous K order to match B):
    //   i in [0,16): K = c*32 + khalf*16 + i
    const float* __restrict__ xrow = x + (size_t)(row0 + r) * DDIM;

    v16bf afrag[KCHUNKS];
    #pragma unroll
    for (int c = 0; c < KCHUNKS; ++c) {
        const int kb = c * 32 + khalf * 16;
        float4 f0 = *(const float4*)(xrow + kb);
        float4 f1 = *(const float4*)(xrow + kb + 4);
        float4 f2 = *(const float4*)(xrow + kb + 8);
        float4 f3 = *(const float4*)(xrow + kb + 12);
        v16bf a;
        a[0]  = (__bf16)f0.x; a[1]  = (__bf16)f0.y; a[2]  = (__bf16)f0.z; a[3]  = (__bf16)f0.w;
        a[4]  = (__bf16)f1.x; a[5]  = (__bf16)f1.y; a[6]  = (__bf16)f1.z; a[7]  = (__bf16)f1.w;
        a[8]  = (__bf16)f2.x; a[9]  = (__bf16)f2.y; a[10] = (__bf16)f2.z; a[11] = (__bf16)f2.w;
        a[12] = (__bf16)f3.x; a[13] = (__bf16)f3.y; a[14] = (__bf16)f3.z; a[15] = (__bf16)f3.w;
        afrag[c] = a;
    }

    // -------- Sweep all centroid column tiles --------
    float bestVal[8];
    int   bestIdx[8];
    #pragma unroll
    for (int v = 0; v < 8; ++v) { bestVal[v] = 3.402823466e38f; bestIdx[v] = 0; }

    for (int jt = 0; jt < COLTILES; ++jt) {
        const int jb = jt * 16;
        const int mycol = jb + r;
        // B fragment (16-bit B 32x16): lane r holds column (jb+r),
        // khalf selects K 0..15 vs 16..31 within the chunk; contiguous in ct.
        const __bf16* __restrict__ bcol =
            ct + (size_t)mycol * DDIM + (size_t)khalf * 16;

        if (jt + 1 < COLTILES) {
            // Keep next tile's B column warm in near caches (RT temporal).
            __builtin_prefetch(ct + (size_t)(mycol + 16) * DDIM, 0, 3);
        }

        // Two independent accumulator chains double WMMA ILP so dependent
        // matrix-op latency is hidden even at low wave occupancy.
        v8f acc0 = {};
        v8f acc1 = {};
        #pragma unroll
        for (int c = 0; c < KCHUNKS; c += 2) {
            v16bf b0 = *(const v16bf*)(bcol + (size_t)c * 32);
            v16bf b1 = *(const v16bf*)(bcol + (size_t)(c + 1) * 32);
            acc0 = __builtin_amdgcn_wmma_f32_16x16x32_bf16(
                false, afrag[c], false, b0, (short)0, acc0, false, false);
            acc1 = __builtin_amdgcn_wmma_f32_16x16x32_bf16(
                false, afrag[c + 1], false, b1, (short)0, acc1, false, false);
        }

        // dist = ||c||^2 - 2 x.c  (||x||^2 dropped: constant per row)
        const float cn = cnorm[mycol];
        #pragma unroll
        for (int v = 0; v < 8; ++v) {
            float dot = acc0[v] + acc1[v];
            float dist = __builtin_fmaf(-2.0f, dot, cn);
            if (dist < bestVal[v]) { bestVal[v] = dist; bestIdx[v] = mycol; }
        }
    }

    // -------- Cross-lane argmin: D-tile row v lives in 16 lanes --------
    #pragma unroll
    for (int off = 1; off < 16; off <<= 1) {
        #pragma unroll
        for (int v = 0; v < 8; ++v) {
            float oVal = __shfl_xor(bestVal[v], off, 32);
            int   oIdx = __shfl_xor(bestIdx[v], off, 32);
            if (oVal < bestVal[v] ||
                (oVal == bestVal[v] && oIdx < bestIdx[v])) {
                bestVal[v] = oVal;
                bestIdx[v] = oIdx;
            }
        }
    }

    if (r == 0) {
        const int rowBase = row0 + khalf * 8;   // vgpr v = row v + 8*khalf
        #pragma unroll
        for (int v = 0; v < 8; ++v) {
            out[rowBase + v] = (float)bestIdx[v];
        }
    }
}

// ---------------------------------------------------------------------------
extern "C" void kernel_launch(void* const* d_in, const int* in_sizes, int n_in,
                              void* d_out, int out_size, void* d_ws, size_t ws_size,
                              hipStream_t stream) {
    const float* x      = (const float*)d_in[0];   // [N, D] f32
    const float* cents  = (const float*)d_in[1];   // [D, K] f32
    const float* cnorm  = (const float*)d_in[2];   // [1, K] f32
    float*       out    = (float*)d_out;           // [N] argmin index as float
    __bf16*      ct     = (__bf16*)d_ws;           // [K, D] bf16 (3 MB scratch)

    // Pass 1: transpose+convert centroids to bf16.
    kmeans_convertT_kernel<<<(KCEN * DDIM + 255) / 256, 256, 0, stream>>>(cents, ct);

    // Pass 2: fused WMMA GEMM + argmin. 8192 row tiles, 4 waves/block.
    kmeans_argmin_kernel<<<(NPTS / 16) / WAVES_PER_BLOCK, WAVES_PER_BLOCK * 32, 0, stream>>>(
        x, ct, cnorm, out);
}
